// SelfAttention_13237089206926
// MI455X (gfx1250) — matrix operand order repaired and checked by reading, not compiled, not run on hardware
//
#include <hip/hip_runtime.h>

typedef __attribute__((ext_vector_type(16))) __bf16 v16bf;
typedef __attribute__((ext_vector_type(8)))  float  v8f;

union FragBF { v16bf v; unsigned int u[8]; uint4 q[2]; };
union FragF  { v8f  v;  float        f[8]; };

#ifndef __has_builtin
#define __has_builtin(x) 0
#endif
#if __has_builtin(__builtin_amdgcn_global_load_async_to_lds_b32) && \
    __has_builtin(__builtin_amdgcn_s_wait_asynccnt)
#define USE_ASYNC 1
#else
#define USE_ASYNC 0
#endif

#if USE_ASYNC
typedef __attribute__((address_space(1))) int gint;
typedef __attribute__((address_space(3))) int lint;
#endif

// async (or fallback) dword copy global -> LDS
__device__ __forceinline__ void cp_b32(const unsigned int* g, unsigned int* l) {
#if USE_ASYNC
  __builtin_amdgcn_global_load_async_to_lds_b32((gint*)g, (lint*)l, 0, 0);
#else
  *l = *g;
#endif
}
__device__ __forceinline__ void cp_wait() {
#if USE_ASYNC
  __builtin_amdgcn_s_wait_asynccnt(0);
#endif
}

__device__ __forceinline__ unsigned int pack2_bf16(float a, float b) {
  unsigned int ua = __float_as_uint(a);
  ua = (ua + 0x7FFFu + ((ua >> 16) & 1u)) >> 16;
  unsigned int ub = __float_as_uint(b);
  ub = (ub + 0x7FFFu + ((ub >> 16) & 1u)) >> 16;
  return (ua & 0xFFFFu) | (ub << 16);
}
__device__ __forceinline__ unsigned short f32_to_bf16u(float a) {
  unsigned int ua = __float_as_uint(a);
  ua = (ua + 0x7FFFu + ((ua >> 16) & 1u)) >> 16;
  return (unsigned short)ua;
}

// fragment-swizzle helpers: map a k-pair index (0..15 within a 32-wide k chunk)
// to (lane-half, vgpr) slots of the documented WMMA bf16 layouts.
__device__ __forceinline__ int swzA_half(int kpl) { return (kpl >> 2) & 1; }
__device__ __forceinline__ int swzA_v(int kpl)    { return (kpl & 3) | ((kpl >> 3) << 2); }
__device__ __forceinline__ int swzB_half(int kpl) { return (kpl >> 2) & 1; }
__device__ __forceinline__ int swzB_v(int kpl)    { return (kpl & 3) | ((kpl >> 3) << 2); }

// ---------------------------------------------------------------------------
// Prep kernels: one-time conversion / packing into WMMA-friendly layouts.
// ---------------------------------------------------------------------------
__global__ __launch_bounds__(256)
void cvt_f32_bf16pk(const float* __restrict__ X, unsigned int* __restrict__ Y, int total) {
  int idx = blockIdx.x * 256 + threadIdx.x;
  if (idx >= total) return;
  Y[idx] = pack2_bf16(X[2 * (size_t)idx], X[2 * (size_t)idx + 1]);
}
// W fp32 [K][N] row-major -> Wp dwords [K/2][N], Wp[kp][n] = pack(W[2kp][n], W[2kp+1][n])
__global__ __launch_bounds__(256)
void pack_weight(const float* __restrict__ W, unsigned int* __restrict__ Wp, int N, int total) {
  int idx = blockIdx.x * 256 + threadIdx.x;
  if (idx >= total) return;
  int kp = idx / N, n = idx - kp * N;
  Wp[idx] = pack2_bf16(W[(size_t)(2 * kp) * N + n], W[(size_t)(2 * kp + 1) * N + n]);
}
// sr_w fp32 [O=768][I=768][8] -> conv B packed dwords [3072][768] with K order off*768+i
__global__ __launch_bounds__(256)
void pack_conv_weight(const float* __restrict__ W, unsigned int* __restrict__ Wp) {
  int idx = blockIdx.x * 256 + threadIdx.x;
  if (idx >= 3072 * 768) return;
  int kp = idx / 768, n = idx - kp * 768;
  int k0 = kp << 1;
  int off = k0 / 768, i = k0 - off * 768;   // i even; pair (i, i+1) shares off
  const float* p = W + (size_t)n * 6144 + (size_t)i * 8 + off;
  Wp[idx] = pack2_bf16(p[0], p[8]);
}

// ---------------------------------------------------------------------------
// Tiled bf16 WMMA GEMM, all operands pre-packed bf16.
// LDS tiles are stored FRAGMENT-SWIZZLED: [16x16-tile][lane][v], so every
// fragment load is 2x ds_load_b128 (contiguous 32B per lane, no padding).
//   A: bf16-pair dwords [M][K/2] (AMODE 0) or im2col gather from x-bf16 (AMODE 1)
//   B: bf16-pair dwords [K/2][N]
// OUTMODE: 0 = fp32 (+bias), 1 = bf16 row-major (scaled by oscale), 2 = split KV
//          (cols <768 -> K bf16 [M][768]; cols >=768 -> V transposed [768][M])
// 128x128 tile, K-step 32, 256 threads, double-buffered async LDS staging.
// ---------------------------------------------------------------------------
template<int AMODE, int OUTMODE, bool BIAS>
__global__ __launch_bounds__(256)
void wmma_gemm(const unsigned int* __restrict__ Au, const unsigned int* __restrict__ Bp,
               const float* __restrict__ bias,
               float* __restrict__ outF, unsigned short* __restrict__ out16,
               unsigned short* __restrict__ outV,
               int M, int N, int K, float oscale)
{
  __shared__ alignas(16) unsigned int Alds[2][2048];   // 8 m-tiles * 32 lanes * 8 v
  __shared__ alignas(16) unsigned int Blds[2][2048];   // 8 n-tiles * 32 lanes * 8 v

  const int tid  = threadIdx.x;
  const int m0   = blockIdx.y * 128;
  const int n0   = blockIdx.x * 128;
  const int lane = tid & 31;
  const int wave = tid >> 5;
  const int half = lane >> 4;
  const int ln   = lane & 15;
  const int wm   = wave >> 1;   // 0..3
  const int wn   = wave & 1;    // 0..1

  auto stage = [&](int kt, int buf) {
    for (int t = tid; t < 128 * 16; t += 256) {          // A tile: 2048 dwords
      int row = t >> 4, kpl = t & 15;
      int gm = m0 + row;
      int gmc = (gm < M) ? gm : 0;                       // clamp: junk rows discarded later
      const unsigned int* src;
      if (AMODE == 0) {
        src = Au + (size_t)gmc * (K >> 1) + (kt << 4) + kpl;
      } else {
        int gk = (kt << 5) + (kpl << 1);
        int off = gk / 768;
        int i   = gk - off * 768;
        int dz = off >> 2, dy = (off >> 1) & 1, dx = off & 1;
        int zo = gmc / 144; int rem = gmc - zo * 144;
        int yo = rem / 12;  int xo = rem - yo * 12;
        int nidx = (2 * zo + dz) * 576 + (2 * yo + dy) * 24 + (2 * xo + dx);
        src = Au + (size_t)nidx * 384 + (i >> 1);
      }
      int dst = (row >> 4) * 256 + (swzA_half(kpl) * 16 + (row & 15)) * 8 + swzA_v(kpl);
      cp_b32(src, &Alds[buf][dst]);
    }
    for (int t = tid; t < 16 * 128; t += 256) {          // B tile: 2048 dwords
      int kpl = t >> 7, nc = t & 127;
      int dst = (nc >> 4) * 256 + (swzB_half(kpl) * 16 + (nc & 15)) * 8 + swzB_v(kpl);
      cp_b32(Bp + (size_t)((kt << 4) + kpl) * N + (n0 + nc), &Blds[buf][dst]);
    }
  };

  FragF acc[2][4];
  for (int mi = 0; mi < 2; mi++)
    for (int nj = 0; nj < 4; nj++)
      for (int r = 0; r < 8; r++) acc[mi][nj].f[r] = 0.0f;

  stage(0, 0);
  cp_wait();
  __syncthreads();

  const int ksteps = K >> 5;
  for (int kt = 0; kt < ksteps; ++kt) {
    const int buf = kt & 1;
    if (kt + 1 < ksteps) stage(kt + 1, buf ^ 1);

    FragBF af[2], bfr[4];
    #pragma unroll
    for (int mi = 0; mi < 2; mi++) {
      const uint4* pa = (const uint4*)&Alds[buf][(wm * 2 + mi) * 256 + lane * 8];
      af[mi].q[0] = pa[0]; af[mi].q[1] = pa[1];
    }
    #pragma unroll
    for (int nj = 0; nj < 4; nj++) {
      const uint4* pb = (const uint4*)&Blds[buf][(wn * 4 + nj) * 256 + lane * 8];
      bfr[nj].q[0] = pb[0]; bfr[nj].q[1] = pb[1];
    }
    #pragma unroll
    for (int mi = 0; mi < 2; mi++)
      #pragma unroll
      for (int nj = 0; nj < 4; nj++)
        acc[mi][nj].v = __builtin_amdgcn_wmma_f32_16x16x32_bf16(
            false, af[mi].v, false, bfr[nj].v, (short)0, acc[mi][nj].v, false, false);

    cp_wait();
    __syncthreads();
  }

  for (int mi = 0; mi < 2; mi++)
    for (int nj = 0; nj < 4; nj++) {
      int gn = n0 + wn * 64 + nj * 16 + ln;
      float badd = BIAS ? bias[gn] : 0.0f;
      #pragma unroll
      for (int r = 0; r < 8; r++) {
        int gm = m0 + wm * 32 + mi * 16 + r + 8 * half;
        if (gm >= M) continue;
        float val = acc[mi][nj].f[r] + badd;
        if (OUTMODE == 0) {
          outF[(size_t)gm * N + gn] = val;
        } else if (OUTMODE == 1) {
          out16[(size_t)gm * N + gn] = f32_to_bf16u(val * oscale);
        } else {
          if (gn < 768) out16[(size_t)gm * 768 + gn] = f32_to_bf16u(val);
          else          outV[(size_t)(gn - 768) * M + gm] = f32_to_bf16u(val);
        }
      }
    }
}

// ---------------------------------------------------------------------------
// LayerNorm over C=768 (fp32 in, packed bf16-pair dwords out), 1 block/row
// ---------------------------------------------------------------------------
__global__ __launch_bounds__(256)
void layernorm768(const float* __restrict__ X, const float* __restrict__ g,
                  const float* __restrict__ b, unsigned int* __restrict__ Y)
{
  __shared__ float red[256];
  const int row = blockIdx.x;
  const int tid = threadIdx.x;
  const float* xr = X + (size_t)row * 768;
  float v0a = xr[2 * tid], v0b = xr[2 * tid + 1];
  float v1a = 0.f, v1b = 0.f;
  if (tid < 128) { v1a = xr[512 + 2 * tid]; v1b = xr[512 + 2 * tid + 1]; }
  float s  = v0a + v0b + v1a + v1b;
  float s2 = v0a * v0a + v0b * v0b + v1a * v1a + v1b * v1b;
  red[tid] = s; __syncthreads();
  for (int o = 128; o > 0; o >>= 1) { if (tid < o) red[tid] += red[tid + o]; __syncthreads(); }
  float mean = red[0] * (1.0f / 768.0f);
  __syncthreads();
  red[tid] = s2; __syncthreads();
  for (int o = 128; o > 0; o >>= 1) { if (tid < o) red[tid] += red[tid + o]; __syncthreads(); }
  float var  = red[0] * (1.0f / 768.0f) - mean * mean;
  float rstd = rsqrtf(var + 1e-5f);
  int c0 = 2 * tid;
  Y[(size_t)row * 384 + tid] =
      pack2_bf16((v0a - mean) * rstd * g[c0] + b[c0],
                 (v0b - mean) * rstd * g[c0 + 1] + b[c0 + 1]);
  if (tid < 128) {
    int c1 = 512 + 2 * tid;
    Y[(size_t)row * 384 + 256 + tid] =
        pack2_bf16((v1a - mean) * rstd * g[c1] + b[c1],
                   (v1b - mean) * rstd * g[c1 + 1] + b[c1 + 1]);
  }
}

// ---------------------------------------------------------------------------
// Flash attention: grid (108 query blocks, 8 heads), 256 threads (8 waves).
// Q bf16 [13824][768] (pre-scaled by 1/sqrt(8)), K bf16 [1728][768],
// V bf16 TRANSPOSED [768][1728]. 54 key blocks of 32, double-buffered async
// staging, fragment-swizzled LDS. Softmax denominator via ones-column WMMA.
// ---------------------------------------------------------------------------
__global__ __launch_bounds__(256)
void attn_flash(const unsigned int* __restrict__ Qu, const unsigned int* __restrict__ Ku,
                const unsigned int* __restrict__ Vu, unsigned short* __restrict__ Obf)
{
  __shared__ alignas(16) unsigned int Qlds[8 * 3 * 256];   // [rowtile][kc][lane][v]
  __shared__ alignas(16) unsigned int Klds[2][2 * 3 * 256];// [keytile][kc][lane][v]
  __shared__ alignas(16) unsigned int Vlds[2][6 * 256];    // [dcoltile][lane][v]
  __shared__ alignas(16) unsigned int PldsU[8 * 256];      // P, A-swizzled per wave

  const int tid  = threadIdx.x;
  const int qb   = blockIdx.x;
  const int h    = blockIdx.y;
  const int lane = tid & 31;
  const int wave = tid >> 5;
  const int half = lane >> 4;
  const int ln   = lane & 15;

  auto stageKV = [&](int kb, int buf) {
    for (int t = tid; t < 32 * 48; t += 256) {   // K: 32 keys x 48 dpairs
      int key = t / 48, dp = t % 48;
      int kc = dp >> 4, kpl = dp & 15;
      int dst = ((key >> 4) * 3 + kc) * 256 + (swzB_half(kpl) * 16 + (key & 15)) * 8 + swzB_v(kpl);
      cp_b32(Ku + (size_t)(kb * 32 + key) * 384 + h * 48 + dp, &Klds[buf][dst]);
    }
    for (int t = tid; t < 96 * 16; t += 256) {   // V: 96 dcols x 16 keypairs (dc-major, coalesced)
      int dc = t >> 4, kpl = t & 15;
      int dst = (dc >> 4) * 256 + (swzB_half(kpl) * 16 + (dc & 15)) * 8 + swzB_v(kpl);
      cp_b32(Vu + (size_t)(h * 96 + dc) * 864 + kb * 16 + kpl, &Vlds[buf][dst]);
    }
  };

  for (int t = tid; t < 128 * 48; t += 256) {    // Q: 128 rows x 48 dpairs
    int row = t / 48, dp = t % 48;
    int kc = dp >> 4, kpl = dp & 15;
    int dst = ((row >> 4) * 3 + kc) * 256 + (swzA_half(kpl) * 16 + (row & 15)) * 8 + swzA_v(kpl);
    cp_b32(Qu + (size_t)(qb * 128 + row) * 384 + h * 48 + dp, &Qlds[dst]);
  }
  stageKV(0, 0);
  cp_wait();
  __syncthreads();

  // acc[0..5] = O accumulator; acc[6] = softmax denominator (ones-column)
  FragF acc[7];
  for (int c = 0; c < 7; c++) for (int r = 0; r < 8; r++) acc[c].f[r] = 0.f;
  float mrow[8];
  for (int r = 0; r < 8; r++) mrow[r] = -1e30f;

  FragBF vone;                                   // constant all-ones bf16 B fragment
  #pragma unroll
  for (int v = 0; v < 8; v++) vone.u[v] = 0x3F803F80u;

  for (int kb = 0; kb < 54; ++kb) {
    const int buf = kb & 1;
    if (kb + 1 < 54) stageKV(kb + 1, buf ^ 1);

    // S = Q * K^T  (16 rows x 32 keys per wave); Q pre-scaled by 1/sqrt(8)
    FragF s[2];
    for (int nj = 0; nj < 2; nj++) for (int r = 0; r < 8; r++) s[nj].f[r] = 0.f;
    #pragma unroll
    for (int kc = 0; kc < 3; kc++) {
      FragBF aq;
      const uint4* pq = (const uint4*)&Qlds[(wave * 3 + kc) * 256 + lane * 8];
      aq.q[0] = pq[0]; aq.q[1] = pq[1];
      #pragma unroll
      for (int nj = 0; nj < 2; nj++) {
        FragBF bk;
        const uint4* pk = (const uint4*)&Klds[buf][(nj * 3 + kc) * 256 + lane * 8];
        bk.q[0] = pk[0]; bk.q[1] = pk[1];
        s[nj].v = __builtin_amdgcn_wmma_f32_16x16x32_bf16(
            false, aq.v, false, bk.v, (short)0, s[nj].v, false, false);
      }
    }

    // online softmax: row m = r + 8*half lives in VGPR r across lanes 0..15.
    // Only the max needs a cross-lane butterfly; the denominator comes from
    // the ones-column WMMA below.
    #pragma unroll
    for (int r = 0; r < 8; r++) {
      float mx = fmaxf(s[0].f[r], s[1].f[r]);
      for (int o = 1; o < 16; o <<= 1) mx = fmaxf(mx, __shfl_xor(mx, o, 32));
      float mnew = fmaxf(mrow[r], mx);
      float corr = __expf(mrow[r] - mnew);
      mrow[r] = mnew;
      s[0].f[r] = __expf(s[0].f[r] - mnew);
      s[1].f[r] = __expf(s[1].f[r] - mnew);
      #pragma unroll
      for (int c = 0; c < 7; c++) acc[c].f[r] *= corr;
    }

    // P: C-layout -> A-swizzled LDS (same-wave region, DS ops in-order)
    unsigned short* Plds = (unsigned short*)PldsU;
    #pragma unroll
    for (int nj = 0; nj < 2; nj++) {
      int kpp   = nj * 8 + (ln >> 1);            // k-pair index of this lane's column
      int bit   = ln & 1;
      int ha    = swzA_half(kpp);
      int va    = swzA_v(kpp);
      #pragma unroll
      for (int r = 0; r < 8; r++) {
        int mloc = r + 8 * half;
        Plds[(wave * 256 + (ha * 16 + mloc) * 8 + va) * 2 + bit] = f32_to_bf16u(s[nj].f[r]);
      }
    }

    // O += P * V ; denom += P * ones  (K-dim = 32 keys, one WMMA k-step)
    {
      FragBF ap;
      const uint4* pp = (const uint4*)&PldsU[wave * 256 + lane * 8];
      ap.q[0] = pp[0]; ap.q[1] = pp[1];
      #pragma unroll
      for (int c = 0; c < 6; c++) {
        FragBF bv;
        const uint4* pv = (const uint4*)&Vlds[buf][c * 256 + lane * 8];
        bv.q[0] = pv[0]; bv.q[1] = pv[1];
        acc[c].v = __builtin_amdgcn_wmma_f32_16x16x32_bf16(
            false, ap.v, false, bv.v, (short)0, acc[c].v, false, false);
      }
      acc[6].v = __builtin_amdgcn_wmma_f32_16x16x32_bf16(
          false, ap.v, false, vone.v, (short)0, acc[6].v, false, false);
    }

    cp_wait();
    __syncthreads();
  }

  #pragma unroll
  for (int r = 0; r < 8; r++) {
    float inv = 1.0f / acc[6].f[r];              // rowsum (same in every lane/column)
    int gm = qb * 128 + wave * 16 + r + 8 * half;
    #pragma unroll
    for (int c = 0; c < 6; c++) {
      int gc = h * 96 + c * 16 + ln;
      Obf[(size_t)gm * 768 + gc] = f32_to_bf16u(acc[c].f[r] * inv);
    }
  }
}

// ---------------------------------------------------------------------------
extern "C" void kernel_launch(void* const* d_in, const int* in_sizes, int n_in,
                              void* d_out, int out_size, void* d_ws, size_t ws_size,
                              hipStream_t stream)
{
  (void)in_sizes; (void)n_in; (void)out_size; (void)ws_size;
  const float* x      = (const float*)d_in[0];
  const float* wq     = (const float*)d_in[1];
  const float* wkv    = (const float*)d_in[2];
  const float* sr_w   = (const float*)d_in[3];
  const float* sr_b   = (const float*)d_in[4];
  const float* ln_g   = (const float*)d_in[5];
  const float* ln_b   = (const float*)d_in[6];
  const float* proj_w = (const float*)d_in[7];
  const float* proj_b = (const float*)d_in[8];
  float* out = (float*)d_out;

  const int N = 13824, C = 768, Nr = 1728;

  char* p = (char*)d_ws;
  auto carve = [&](size_t bytes) {
    char* r = p; p += (bytes + 255) & ~(size_t)255; return r;
  };
  unsigned int*   xbf   = (unsigned int*)  carve((size_t)N * 384 * 4);     // x as bf16 pairs
  unsigned int*   wqp   = (unsigned int*)  carve((size_t)384 * 768 * 4);
  unsigned int*   wkvp  = (unsigned int*)  carve((size_t)384 * 1536 * 4);
  unsigned int*   prjp  = (unsigned int*)  carve((size_t)384 * 768 * 4);
  unsigned int*   cvp   = (unsigned int*)  carve((size_t)3072 * 768 * 4);  // conv B packed
  unsigned short* Qbf   = (unsigned short*)carve((size_t)N * C * 2);
  float*          xr    = (float*)         carve((size_t)Nr * C * 4);
  unsigned int*   xlnbf = (unsigned int*)  carve((size_t)Nr * 384 * 4);
  unsigned short* Kbf   = (unsigned short*)carve((size_t)Nr * C * 2);
  unsigned short* Vt    = (unsigned short*)carve((size_t)C * Nr * 2);      // V transposed
  unsigned short* Abf   = (unsigned short*)carve((size_t)N * C * 2);       // attn out bf16

  // ---- one-time packing ----
  cvt_f32_bf16pk<<<dim3((N * 384 + 255) / 256), 256, 0, stream>>>(x, xbf, N * 384);
  pack_weight<<<dim3((384 * 768 + 255) / 256), 256, 0, stream>>>(wq, wqp, 768, 384 * 768);
  pack_weight<<<dim3((384 * 1536 + 255) / 256), 256, 0, stream>>>(wkv, wkvp, 1536, 384 * 1536);
  pack_weight<<<dim3((384 * 768 + 255) / 256), 256, 0, stream>>>(proj_w, prjp, 768, 384 * 768);
  pack_conv_weight<<<dim3((3072 * 768 + 255) / 256), 256, 0, stream>>>(sr_w, cvp);

  // 1) Q = (x @ wq) * (1/sqrt(H))  (bf16 out, pre-scaled for attention)
  wmma_gemm<0, 1, false><<<dim3(6, 108), 256, 0, stream>>>(
      xbf, wqp, nullptr, nullptr, Qbf, nullptr, N, C, C, 0.35355339059327373f);
  // 2) spatial-reduction conv as im2col GEMM (+bias), fp32 out for LN stats
  wmma_gemm<1, 0, true><<<dim3(6, 14), 256, 0, stream>>>(
      xbf, cvp, sr_b, xr, nullptr, nullptr, Nr, C, 6144, 1.0f);
  // 3) LayerNorm -> packed bf16
  layernorm768<<<dim3(Nr), 256, 0, stream>>>(xr, ln_g, ln_b, xlnbf);
  // 4) KV GEMM -> K row-major bf16 + V transposed bf16
  wmma_gemm<0, 2, false><<<dim3(12, 14), 256, 0, stream>>>(
      xlnbf, wkvp, nullptr, nullptr, Kbf, Vt, Nr, 2 * C, C, 1.0f);
  // 5) flash attention -> bf16
  attn_flash<<<dim3(108, 8), 256, 0, stream>>>(
      (const unsigned int*)Qbf, (const unsigned int*)Kbf, (const unsigned int*)Vt, Abf);
  // 6) projection + bias -> fp32 output
  wmma_gemm<0, 0, true><<<dim3(6, 108), 256, 0, stream>>>(
      (const unsigned int*)Abf, prjp, proj_b, out, nullptr, nullptr, N, C, C, 1.0f);
}
